// CrossAttentionReranker_74586402062610
// MI455X (gfx1250) — matrix operand order repaired and checked by reading, not compiled, not run on hardware
//
#include <hip/hip_runtime.h>

#define N_CAND 131072
#define DIM    512
#define HID    256
#define ROWS   64          // rows per block, 4 waves x 16-row WMMA tiles
#define PX     (DIM + 8)   // LDS pitch (halves) for K=512 tiles (TDM pads 16B/row)
#define PW     (DIM + 8)
#define PH     (HID + 8)   // LDS pitch for K=256 tiles
#define LN_EPS 1e-5f

typedef __attribute__((ext_vector_type(16))) __bf16 v16bf;
typedef __attribute__((ext_vector_type(8)))  __bf16 v8bf;
typedef __attribute__((ext_vector_type(8)))  float  v8f;
typedef __attribute__((ext_vector_type(4)))  unsigned int u32x4;
typedef __attribute__((ext_vector_type(8)))  unsigned int u32x8;

union BF16x16 { v16bf v; v8bf h[2]; };

// ---------------------------------------------------------------------------
// Tensor Data Mover: DMA a [rows x K] bf16 row-major tile from global memory
// into LDS, padding each row by 16B so the LDS pitch is K+8 halves (bank spread).
// D# per CDNA5 ISA: group0 {count,lds,gaddr,type=2},
// group1 {data_size=2B, pad_enable, pad_interval, pad_amount=4DW, dims, stride}.
// Issued by one wave; completion tracked with TENSORcnt.
__device__ __forceinline__ void tdm_load_tile(unsigned lds_addr, unsigned long gaddr,
                                              int K, int rows, int padcode)
{
    u32x4 g0;
    g0[0] = 1u;                                            // count=1, user descriptor
    g0[1] = lds_addr;                                      // LDS byte address
    g0[2] = (unsigned)gaddr;                               // global_addr[31:0]
    g0[3] = ((unsigned)(gaddr >> 32) & 0x1FFFFFFu) | (2u << 30);  // [56:32] | type=2
    u32x8 g1;
    g1[0] = (1u << 16)                                     // data_size = 1 -> 2 bytes
          | (1u << 20)                                     // pad_enable
          | ((unsigned)padcode << 22)                      // pad_interval (row DWORDs)
          | (3u << 25);                                    // pad_amount = 4 DWORDs (16B)
    g1[1] = ((unsigned)K & 0xFFFFu) << 16;                 // tensor_dim0 low16
    g1[2] = ((unsigned)K >> 16) | ((unsigned)rows << 16);  // tensor_dim0 hi / dim1 lo
    g1[3] = ((unsigned)K << 16);                           // dim1 hi=0 | tile_dim0=K
    g1[4] = (unsigned)rows;                                // tile_dim1 | tile_dim2=0
    g1[5] = (unsigned)K;                                   // tensor_dim0_stride low32
    g1[6] = 0u;                                            // stride hi | dim1_stride lo
    g1[7] = 0u;
    asm volatile("tensor_load_to_lds %0, %1" :: "s"(g0), "s"(g1) : "memory");
}

__device__ __forceinline__ unsigned lds_addr_of(const void* p)
{
    // flat LDS pointers carry the LDS byte offset in the low 32 bits (ISA 10.2)
    return (unsigned)(unsigned long)p;
}

// One 16x16 D tile: A = LDS activations [16 rows x K], B = LDS weights [16 outs x K].
// A-frag: lane m=lane&15 -> row m; halves 0..7 = K(k+hi*8 .. +7), 8..15 = K(+16).
// B-frag: lane n=lane&15 -> weight row n; halves 0..7 = K(k+hi*8 .. +7), 8..15 = K(+16).
__device__ __forceinline__ v8f wmma_rowtile(const __bf16* Xl, int pitchX, int rowbase,
                                            const __bf16* Wl, int pitchW, int K, v8f acc)
{
    int lane = threadIdx.x & 31;
    int m    = lane & 15;
    int hi   = lane >> 4;
    const __bf16* xrow = Xl + (rowbase + m) * pitchX;
    const __bf16* wrow = Wl + m * pitchW;
#pragma unroll
    for (int k = 0; k < K; k += 32) {
        BF16x16 a, b;
        a.h[0] = *(const v8bf*)(xrow + k + hi * 8);
        a.h[1] = *(const v8bf*)(xrow + k + hi * 8 + 16);
        b.h[0] = *(const v8bf*)(wrow + k + hi * 8);
        b.h[1] = *(const v8bf*)(wrow + k + hi * 8 + 16);
        acc = __builtin_amdgcn_wmma_f32_16x16x32_bf16(false, a.v, false, b.v,
                                                      (short)0, acc, false, false);
    }
    return acc;
}

// Scatter a 16x16 f32 D tile into an LDS bf16 tile (+bias, optional relu).
__device__ __forceinline__ void store_tile(__bf16* Tl, int pitchT, int rowbase, int colbase,
                                           v8f acc, float bias, bool relu)
{
    int lane = threadIdx.x & 31;
    int col  = colbase + (lane & 15);
    int r0   = rowbase + (lane >> 4) * 8;
#pragma unroll
    for (int r = 0; r < 8; ++r) {
        float v = acc[r] + bias;
        if (relu) v = fmaxf(v, 0.f);
        Tl[(r0 + r) * pitchT + col] = (__bf16)v;
    }
}

// dst[row] = LN(resid[row] + Yl[row] + bias) * lnw + lnb   (per-wave rows, f32 math)
__device__ __forceinline__ void add_ln_store(const __bf16* Yl, int pitchY,
                                             const __bf16* resid, __bf16* dst, long rowstart,
                                             const float* __restrict__ bias,
                                             const float* __restrict__ lnw,
                                             const float* __restrict__ lnb)
{
    int wave = threadIdx.x >> 5, lane = threadIdx.x & 31;
    float bj[16], wj[16], oj[16];
#pragma unroll
    for (int j = 0; j < 16; ++j) {
        int col = lane + j * 32;
        bj[j] = bias[col]; wj[j] = lnw[col]; oj[j] = lnb[col];
    }
    for (int r = 0; r < 16; ++r) {
        int row = wave * 16 + r;
        const __bf16* y   = Yl + row * pitchY;
        const __bf16* res = resid + (rowstart + row) * (long)DIM;
        float v[16], s1 = 0.f, s2 = 0.f;
#pragma unroll
        for (int j = 0; j < 16; ++j) {
            int col = lane + j * 32;
            float t = (float)y[col] + (float)res[col] + bj[j];
            v[j] = t; s1 += t; s2 += t * t;
        }
#pragma unroll
        for (int off = 16; off; off >>= 1) {
            s1 += __shfl_xor(s1, off, 32);
            s2 += __shfl_xor(s2, off, 32);
        }
        float mu = s1 * (1.f / DIM);
        float rs = rsqrtf(s2 * (1.f / DIM) - mu * mu + LN_EPS);
        __bf16* o = dst + (rowstart + row) * (long)DIM;
#pragma unroll
        for (int j = 0; j < 16; ++j)
            o[lane + j * 32] = (__bf16)((v[j] - mu) * rs * wj[j] + oj[j]);
    }
}

// ---------------------------------------------------------------------------
// dst = LN(resid + (src @ wv^T + bv) @ ow^T + ob)   -- fused "MHA" + LayerNorm
// Weight tiles DMA'd by the TDM (single-buffered: LDS budget), activations too.
__global__ void __launch_bounds__(128)
mha_ln_kernel(const __bf16* src, const __bf16* resid, __bf16* dst,
              const __bf16* __restrict__ wv, const float* __restrict__ bv,
              const __bf16* __restrict__ ow, const float* __restrict__ ob,
              const float* __restrict__ lnw, const float* __restrict__ lnb)
{
    __shared__ __attribute__((aligned(16))) __bf16 Xl[ROWS * PX];  // src tile, then y2
    __shared__ __attribute__((aligned(16))) __bf16 Tl[ROWS * PX];  // t1 tile
    __shared__ __attribute__((aligned(16))) __bf16 Wl[16 * PW];
    int tid = threadIdx.x, wave = tid >> 5, lane = tid & 31;
    long rowstart = (long)blockIdx.x * ROWS;
    unsigned ldsX = lds_addr_of(Xl), ldsW = lds_addr_of(Wl);

    if (wave == 0) {
        tdm_load_tile(ldsX, (unsigned long)(src + rowstart * DIM), DIM, ROWS, 7);
        tdm_load_tile(ldsW, (unsigned long)wv, DIM, 16, 7);
    }
    for (int ot = 0; ot < DIM / 16; ++ot) {                 // t1 = src @ wv^T + bv
        if (wave == 0) __builtin_amdgcn_s_wait_tensorcnt(0);
        __syncthreads();
        v8f acc = {0, 0, 0, 0, 0, 0, 0, 0};
        acc = wmma_rowtile(Xl, PX, wave * 16, Wl, PW, DIM, acc);
        store_tile(Tl, PX, wave * 16, ot * 16, acc, bv[ot * 16 + (lane & 15)], false);
        __syncthreads();
        if (wave == 0 && ot + 1 < DIM / 16)
            tdm_load_tile(ldsW, (unsigned long)(wv + (long)(ot + 1) * 16 * DIM), DIM, 16, 7);
    }
    if (wave == 0)
        tdm_load_tile(ldsW, (unsigned long)ow, DIM, 16, 7);
    for (int ot = 0; ot < DIM / 16; ++ot) {                 // y2 = t1 @ ow^T  (ob in LN)
        if (wave == 0) __builtin_amdgcn_s_wait_tensorcnt(0);
        __syncthreads();
        v8f acc = {0, 0, 0, 0, 0, 0, 0, 0};
        acc = wmma_rowtile(Tl, PX, wave * 16, Wl, PW, DIM, acc);
        store_tile(Xl, PX, wave * 16, ot * 16, acc, 0.f, false);
        __syncthreads();
        if (wave == 0 && ot + 1 < DIM / 16)
            tdm_load_tile(ldsW, (unsigned long)(ow + (long)(ot + 1) * 16 * DIM), DIM, 16, 7);
    }
    add_ln_store(Xl, PX, resid, dst, rowstart, ob, lnw, lnb);
}

// dst = LN(x + relu(x @ w1^T + b1) @ w2^T + b2)      -- fused FFN + LayerNorm
// Weight tiles double-buffered: TDM prefetches tile ot+2 while computing ot.
__global__ void __launch_bounds__(128)
ffn_ln_kernel(const __bf16* x, __bf16* dst,
              const __bf16* __restrict__ w1, const float* __restrict__ b1,
              const __bf16* __restrict__ w2, const float* __restrict__ b2,
              const float* __restrict__ lnw, const float* __restrict__ lnb)
{
    __shared__ __attribute__((aligned(16))) __bf16 Xl[ROWS * PX];  // x tile, then y2
    __shared__ __attribute__((aligned(16))) __bf16 Hl[ROWS * PH];  // hidden tile
    __shared__ __attribute__((aligned(16))) __bf16 Wl[2 * 16 * PW];
    int tid = threadIdx.x, wave = tid >> 5, lane = tid & 31;
    long rowstart = (long)blockIdx.x * ROWS;
    unsigned ldsX = lds_addr_of(Xl), ldsW = lds_addr_of(Wl);

    if (wave == 0) {
        tdm_load_tile(ldsX, (unsigned long)(x + rowstart * DIM), DIM, ROWS, 7);
        tdm_load_tile(ldsW,              (unsigned long)w1,                  DIM, 16, 7);
        tdm_load_tile(ldsW + 16 * PW * 2, (unsigned long)(w1 + 16 * DIM),    DIM, 16, 7);
    }
    const int NT1 = HID / 16;
    for (int ot = 0; ot < NT1; ++ot) {                      // h = relu(x @ w1^T + b1)
        if (wave == 0) {
            if (ot < NT1 - 1) __builtin_amdgcn_s_wait_tensorcnt(1);
            else              __builtin_amdgcn_s_wait_tensorcnt(0);
        }
        __syncthreads();
        const __bf16* Wb = Wl + (ot & 1) * 16 * PW;
        v8f acc = {0, 0, 0, 0, 0, 0, 0, 0};
        acc = wmma_rowtile(Xl, PX, wave * 16, Wb, PW, DIM, acc);
        store_tile(Hl, PH, wave * 16, ot * 16, acc, b1[ot * 16 + (lane & 15)], true);
        __syncthreads();
        if (wave == 0 && ot + 2 < NT1)
            tdm_load_tile(ldsW + (ot & 1) * 16 * PW * 2,
                          (unsigned long)(w1 + (long)(ot + 2) * 16 * DIM), DIM, 16, 7);
    }
    if (wave == 0) {
        tdm_load_tile(ldsW,              (unsigned long)w2,                 HID, 16, 6);
        tdm_load_tile(ldsW + 16 * PH * 2, (unsigned long)(w2 + 16 * HID),   HID, 16, 6);
    }
    const int NT2 = DIM / 16;
    for (int ot = 0; ot < NT2; ++ot) {                      // y2 = h @ w2^T  (b2 in LN)
        if (wave == 0) {
            if (ot < NT2 - 1) __builtin_amdgcn_s_wait_tensorcnt(1);
            else              __builtin_amdgcn_s_wait_tensorcnt(0);
        }
        __syncthreads();
        const __bf16* Wb = Wl + (ot & 1) * 16 * PH;
        v8f acc = {0, 0, 0, 0, 0, 0, 0, 0};
        acc = wmma_rowtile(Hl, PH, wave * 16, Wb, PH, HID, acc);
        store_tile(Xl, PX, wave * 16, ot * 16, acc, 0.f, false);
        __syncthreads();
        if (wave == 0 && ot + 2 < NT2)
            tdm_load_tile(ldsW + (ot & 1) * 16 * PH * 2,
                          (unsigned long)(w2 + (long)(ot + 2) * 16 * HID), HID, 16, 6);
    }
    add_ln_store(Xl, PX, x, dst, rowstart, b2, lnw, lnb);
}

// Head GEMM over one 512-wide half of concat([q,c]); w1h is the pre-converted
// bf16 half (256x512 contiguous). phase 0: hacc = partial + b1; phase 1: relu -> h.
__global__ void __launch_bounds__(128)
headA_kernel(const __bf16* x, const __bf16* __restrict__ w1h, const float* __restrict__ b1,
             float* hacc, __bf16* hout, int phase)
{
    __shared__ __attribute__((aligned(16))) __bf16 Xl[ROWS * PX];
    __shared__ __attribute__((aligned(16))) __bf16 Wl[2 * 16 * PW];
    int tid = threadIdx.x, wave = tid >> 5, lane = tid & 31;
    long rowstart = (long)blockIdx.x * ROWS;
    unsigned ldsX = lds_addr_of(Xl), ldsW = lds_addr_of(Wl);

    if (wave == 0) {
        tdm_load_tile(ldsX, (unsigned long)(x + rowstart * DIM), DIM, ROWS, 7);
        tdm_load_tile(ldsW,              (unsigned long)w1h,               DIM, 16, 7);
        tdm_load_tile(ldsW + 16 * PW * 2, (unsigned long)(w1h + 16 * DIM), DIM, 16, 7);
    }
    const int NT = HID / 16;
    for (int ot = 0; ot < NT; ++ot) {
        if (wave == 0) {
            if (ot < NT - 1) __builtin_amdgcn_s_wait_tensorcnt(1);
            else             __builtin_amdgcn_s_wait_tensorcnt(0);
        }
        __syncthreads();
        const __bf16* Wb = Wl + (ot & 1) * 16 * PW;
        v8f acc = {0, 0, 0, 0, 0, 0, 0, 0};
        acc = wmma_rowtile(Xl, PX, wave * 16, Wb, PW, DIM, acc);
        int col = ot * 16 + (lane & 15);
        int r0  = wave * 16 + (lane >> 4) * 8;
#pragma unroll
        for (int r = 0; r < 8; ++r) {
            long row = rowstart + r0 + r;
            if (phase == 0) {
                hacc[row * HID + col] = acc[r] + b1[col];
            } else {
                float t = hacc[row * HID + col] + acc[r];
                hout[row * HID + col] = (__bf16)fmaxf(t, 0.f);
            }
        }
        __syncthreads();
        if (wave == 0 && ot + 2 < NT)
            tdm_load_tile(ldsW + (ot & 1) * 16 * PW * 2,
                          (unsigned long)(w1h + (long)(ot + 2) * 16 * DIM), DIM, 16, 7);
    }
}

// out = sigmoid(h . w2 + b2), one dot of length 256 per row
__global__ void __launch_bounds__(256)
headB_kernel(const __bf16* __restrict__ h, const float* __restrict__ w2,
             const float* __restrict__ b2, float* __restrict__ out)
{
    int wave = threadIdx.x >> 5, lane = threadIdx.x & 31;
    long rowstart = (long)blockIdx.x * 64;
    float bb = b2[0];
    for (int r = 0; r < 8; ++r) {
        long row = rowstart + wave * 8 + r;
        const __bf16* hr = h + row * HID;
        float p = 0.f;
#pragma unroll
        for (int j = 0; j < 8; ++j) {
            int col = lane + j * 32;
            p += (float)hr[col] * w2[col];
        }
#pragma unroll
        for (int off = 16; off; off >>= 1) p += __shfl_xor(p, off, 32);
        if (lane == 0) out[row] = 1.f / (1.f + expf(-(p + bb)));
    }
}

__global__ void __launch_bounds__(256)
init_kernel(const float* __restrict__ query, const float* __restrict__ cand,
            __bf16* q, __bf16* c)
{
    long i = (long)blockIdx.x * blockDim.x + threadIdx.x;
    q[i] = (__bf16)query[i & (DIM - 1)];
    c[i] = (__bf16)cand[i];
}

// Convert an f32 [rows x K] matrix (row stride `rowstride`) to packed bf16 row-major.
__global__ void __launch_bounds__(256)
convw_kernel(const float* __restrict__ src, __bf16* __restrict__ dst,
             int rows, int K, int rowstride)
{
    long i = (long)blockIdx.x * blockDim.x + threadIdx.x;
    if (i < (long)rows * K) {
        int r = (int)(i / K), c = (int)(i % K);
        dst[i] = (__bf16)src[(long)r * rowstride + c];
    }
}

// ---------------------------------------------------------------------------
extern "C" void kernel_launch(void* const* d_in, const int* in_sizes, int n_in,
                              void* d_out, int out_size, void* d_ws, size_t ws_size,
                              hipStream_t stream)
{
    const float* query      = (const float*)d_in[0];
    const float* cand       = (const float*)d_in[1];
    const float* attn_in_w  = (const float*)d_in[2];
    const float* attn_in_b  = (const float*)d_in[3];
    const float* attn_out_w = (const float*)d_in[4];
    const float* attn_out_b = (const float*)d_in[5];
    const float* ln_w       = (const float*)d_in[6];
    const float* ln_b       = (const float*)d_in[7];
    const float* ffn_w1     = (const float*)d_in[8];
    const float* ffn_b1     = (const float*)d_in[9];
    const float* ffn_w2     = (const float*)d_in[10];
    const float* ffn_b2     = (const float*)d_in[11];
    const float* head_w1    = (const float*)d_in[12];
    const float* head_b1    = (const float*)d_in[13];
    const float* head_w2    = (const float*)d_in[14];
    const float* head_b2    = (const float*)d_in[15];
    float* out = (float*)d_out;

    char*   ws   = (char*)d_ws;
    __bf16* q    = (__bf16*)ws;                                     // N*D bf16
    __bf16* c    = (__bf16*)(ws + (size_t)N_CAND * DIM * 2);        // N*D bf16
    float*  hacc = (float*)(ws + (size_t)N_CAND * DIM * 4);         // N*HID f32
    __bf16* wa   = (__bf16*)(ws + (size_t)N_CAND * DIM * 4
                                + (size_t)N_CAND * HID * 4);        // bf16 weight arena
    __bf16* h    = q;   // q is dead after head phase 0; reuse for bf16 hidden

    const size_t SQ = (size_t)DIM * DIM;       // 262144
    const size_t SH = (size_t)HID * DIM;       // 131072
    const size_t SL = 2 * SQ + 2 * SH;         // per-layer arena stride
    __bf16* wv_a[2], * ow_a[2], * w1_a[2], * w2_a[2];
    for (int l = 0; l < 2; ++l) {
        wv_a[l] = wa + l * SL;
        ow_a[l] = wa + l * SL + SQ;
        w1_a[l] = wa + l * SL + 2 * SQ;
        w2_a[l] = wa + l * SL + 2 * SQ + SH;
    }
    __bf16* hw1a = wa + 2 * SL;
    __bf16* hw1b = wa + 2 * SL + SH;

    const int NB = N_CAND / ROWS;   // 2048 row-blocks

    // one-time weight conversion to bf16 (stays L2-resident: ~3.5MB total)
    for (int l = 0; l < 2; ++l) {
        convw_kernel<<<(int)(SQ / 256), 256, 0, stream>>>(
            attn_in_w + (size_t)l * 3 * SQ + 2 * SQ, wv_a[l], DIM, DIM, DIM);
        convw_kernel<<<(int)(SQ / 256), 256, 0, stream>>>(
            attn_out_w + (size_t)l * SQ, ow_a[l], DIM, DIM, DIM);
        convw_kernel<<<(int)(SH / 256), 256, 0, stream>>>(
            ffn_w1 + (size_t)l * SH, w1_a[l], HID, DIM, DIM);
        convw_kernel<<<(int)(SH / 256), 256, 0, stream>>>(
            ffn_w2 + (size_t)l * SH, w2_a[l], DIM, HID, HID);
    }
    convw_kernel<<<(int)(SH / 256), 256, 0, stream>>>(head_w1,       hw1a, HID, DIM, 2 * DIM);
    convw_kernel<<<(int)(SH / 256), 256, 0, stream>>>(head_w1 + DIM, hw1b, HID, DIM, 2 * DIM);

    init_kernel<<<(N_CAND * (long)DIM) / 256, 256, 0, stream>>>(query, cand, q, c);

    for (int l = 0; l < 2; ++l) {
        const float* bv = attn_in_b + (size_t)l * 3 * DIM + 2 * DIM;
        const float* ob = attn_out_b + (size_t)l * DIM;
        const float* lw = ln_w + (size_t)l * DIM;
        const float* lb = ln_b + (size_t)l * DIM;
        const float* b1 = ffn_b1 + (size_t)l * HID;
        const float* b2 = ffn_b2 + (size_t)l * DIM;

        mha_ln_kernel<<<NB, 128, 0, stream>>>(c, q, q, wv_a[l], bv, ow_a[l], ob, lw, lb);
        ffn_ln_kernel<<<NB, 128, 0, stream>>>(q, q, w1_a[l], b1, w2_a[l], b2, lw, lb);
        mha_ln_kernel<<<NB, 128, 0, stream>>>(q, c, c, wv_a[l], bv, ow_a[l], ob, lw, lb);
        ffn_ln_kernel<<<NB, 128, 0, stream>>>(c, c, w1_a[l], b1, w2_a[l], b2, lw, lb);
    }

    // head: [q,c] @ w1^T in two K=512 accumulate passes, then sigmoid(h.w2+b2)
    headA_kernel<<<NB, 128, 0, stream>>>(q, hw1a, head_b1, hacc, h, 0);
    headA_kernel<<<NB, 128, 0, stream>>>(c, hw1b, head_b1, hacc, h, 1);
    headB_kernel<<<NB, 256, 0, stream>>>(h, head_w2, head_b2, out);
}
